// HyperbolicMultiHeadAttention_54400055771511
// MI455X (gfx1250) — matrix operand order repaired and checked by reading, not compiled, not run on hardware
//
#include <hip/hip_runtime.h>
#include <hip/hip_bf16.h>
#include <math.h>

// ---- problem constants (match reference) ----
#define RCC 1.0f        // sqrt(c)
#define PROJ_EPS 4e-3f
#define NHEADS 16
#define BB 2
#define SS 1024
#define DD 1024
#define DH 64

typedef __attribute__((ext_vector_type(16))) __bf16 v16bf;
typedef __attribute__((ext_vector_type(8)))  float  v8f;
typedef __attribute__((ext_vector_type(4)))  int    v4i;

union Frag32 { uint4 q[2]; v16bf b; };   // 32B WMMA A/B fragment

// ---- CDNA5 async global->LDS copy (ASYNCcnt-tracked), with safe fallback ----
#if defined(__HIP_DEVICE_COMPILE__) && __has_builtin(__builtin_amdgcn_global_load_async_to_lds_b128)
#define HAVE_ASYNC_LDS 1
#else
#define HAVE_ASYNC_LDS 0
#endif

typedef __attribute__((address_space(1))) v4i* gv4i_p;   // global 128-bit chunk
typedef __attribute__((address_space(3))) v4i* lv4i_p;   // LDS 128-bit chunk

__device__ __forceinline__ void async_copy16(void* l, const void* g) {
#if HAVE_ASYNC_LDS
  __builtin_amdgcn_global_load_async_to_lds_b128((gv4i_p)(g), (lv4i_p)(l), 0, 0);
#else
  *(uint4*)l = *(const uint4*)g;
#endif
}
__device__ __forceinline__ void async_join() {
#if HAVE_ASYNC_LDS
#if __has_builtin(__builtin_amdgcn_s_wait_asynccnt)
  __builtin_amdgcn_s_wait_asynccnt(0);
#else
  asm volatile("s_wait_asynccnt 0" ::: "memory");
#endif
#endif
}

__device__ __forceinline__ unsigned short f2bf(float f) {
  unsigned u = __float_as_uint(f);
  return (unsigned short)((u + 0x7fffu + ((u >> 16) & 1u)) >> 16);  // RNE
}
__device__ __forceinline__ float rmax16(float v) {
#pragma unroll
  for (int o = 8; o; o >>= 1) v = fmaxf(v, __shfl_xor(v, o, 16));
  return v;
}
__device__ __forceinline__ float rsum16(float v) {
#pragma unroll
  for (int o = 8; o; o >>= 1) v += __shfl_xor(v, o, 16);
  return v;
}

// ---------------------------------------------------------------------------
// K1: per-column z norms, bf16 z_unit, cosh/sinh(2*r)
// ---------------------------------------------------------------------------
__global__ __launch_bounds__(256)
void prep_weights(const float* __restrict__ qz, const float* __restrict__ qr,
                  const float* __restrict__ kz, const float* __restrict__ kr,
                  const float* __restrict__ vz, const float* __restrict__ vr,
                  unsigned short* __restrict__ zub,   // [3][D*D] bf16
                  float* __restrict__ zn, float* __restrict__ ch,
                  float* __restrict__ sh)             // each [3][D]
{
  const int p = blockIdx.x / DD;
  const int j = blockIdx.x % DD;
  const float* z = (p == 0) ? qz : ((p == 1) ? kz : vz);
  const float* r = (p == 0) ? qr : ((p == 1) ? kr : vr);
  __shared__ float red[256];
  float s = 0.f;
  for (int i = threadIdx.x; i < DD; i += 256) { float t = z[(size_t)i * DD + j]; s += t * t; }
  red[threadIdx.x] = s;
  __syncthreads();
  for (int o = 128; o; o >>= 1) {
    if (threadIdx.x < o) red[threadIdx.x] += red[threadIdx.x + o];
    __syncthreads();
  }
  const float norm = fmaxf(sqrtf(red[0]), 1e-15f);
  const float inv = 1.f / norm;
  unsigned short* zu = zub + (size_t)p * DD * DD;
  for (int i = threadIdx.x; i < DD; i += 256)
    zu[(size_t)i * DD + j] = f2bf(z[(size_t)i * DD + j] * inv);
  if (threadIdx.x == 0) {
    zn[p * DD + j] = norm;
    const float drcr = 2.f * RCC * r[j];
    ch[p * DD + j] = coshf(drcr);
    sh[p * DD + j] = sinhf(drcr);
  }
}

// ---------------------------------------------------------------------------
// K2: per-row c*||x||^2, bf16 x
// ---------------------------------------------------------------------------
__global__ __launch_bounds__(256)
void prep_x(const float* __restrict__ x, unsigned short* __restrict__ xb,
            float* __restrict__ cx2)
{
  const int row = blockIdx.x;
  const float* xr = x + (size_t)row * DD;
  unsigned short* xo = xb + (size_t)row * DD;
  __shared__ float red[256];
  float s = 0.f;
  for (int i = threadIdx.x; i < DD; i += 256) { float t = xr[i]; s += t * t; xo[i] = f2bf(t); }
  red[threadIdx.x] = s;
  __syncthreads();
  for (int o = 128; o; o >>= 1) {
    if (threadIdx.x < o) red[threadIdx.x] += red[threadIdx.x + o];
    __syncthreads();
  }
  if (threadIdx.x == 0) cx2[row] = RCC * RCC * red[0];
}

// ---------------------------------------------------------------------------
// K3: bf16 WMMA GEMM [2048x1024]@[1024x1024] + Poincare-MLR epilogue -> yraw
// block: 256 thr (8 waves), tile 128M x 64N, k-step 32, double-buffered async LDS
// ---------------------------------------------------------------------------
__global__ __launch_bounds__(256)
void proj_gemm(const unsigned short* __restrict__ xb,
               const unsigned short* __restrict__ zu,
               const float* __restrict__ zn, const float* __restrict__ ch,
               const float* __restrict__ sh, const float* __restrict__ cx2,
               float* __restrict__ yraw)
{
  __shared__ __align__(32) unsigned short As[2][128 * 32];   // [buf][row][k]
  __shared__ __align__(32) unsigned short Bs[2][32 * 64];    // [buf][k][col]
  const int mb = blockIdx.y, nb = blockIdx.x;
  const int tid = threadIdx.x, w = tid >> 5, lane = tid & 31;
  const int h8 = lane >> 4, nl = lane & 15;

  auto stage = [&](int k0, int buf) {
#pragma unroll
    for (int c = 0; c < 2; ++c) {                            // A tile: 512x16B chunks
      const int cc = tid + c * 256;
      const int ar = cc >> 2, ac = cc & 3;
      async_copy16(&As[buf][ar * 32 + ac * 8],
                   &xb[(size_t)(mb * 128 + ar) * DD + k0 + ac * 8]);
    }
    const int br = tid >> 3, bc = tid & 7;                   // B tile: 256x16B chunks
    async_copy16(&Bs[buf][br * 64 + bc * 8],
                 &zu[(size_t)(k0 + br) * DD + nb * 64 + bc * 8]);
  };

  stage(0, 0);
  async_join();
  __syncthreads();

  v8f acc[4] = {};
  for (int t = 0; t < DD / 32; ++t) {
    const int buf = t & 1;
    if (t + 1 < DD / 32) stage((t + 1) * 32, buf ^ 1);       // prefetch next tile (async)
    Frag32 a;                                                // A: row = w*16+nl, K half by h8
    a.q[0] = *(const uint4*)&As[buf][(w * 16 + nl) * 32 + h8 * 8];
    a.q[1] = *(const uint4*)&As[buf][(w * 16 + nl) * 32 + h8 * 8 + 16];
#pragma unroll
    for (int c2 = 0; c2 < 4; ++c2) {                         // B: col = nl, K half by h8
      v16bf bf;
#pragma unroll
      for (int s = 0; s < 16; ++s) {
        const int kx = (s < 8) ? h8 * 8 + s : 16 + h8 * 8 + (s - 8);
        bf[s] = __builtin_bit_cast(__bf16, Bs[buf][kx * 64 + c2 * 16 + nl]);
      }
      acc[c2] = __builtin_amdgcn_wmma_f32_16x16x32_bf16(
          false, a.b, false, bf, (short)0, acc[c2], false, false);
    }
    async_join();                                            // next tile landed
    __syncthreads();
  }

  // epilogue: w = 2*zn*asinh((2*dot*cosh - (1+cx2)*sinh)/max(1-cx2,eps)); y=sinh(w)
  float cr[8];
#pragma unroll
  for (int r = 0; r < 8; ++r) cr[r] = cx2[mb * 128 + w * 16 + r + 8 * h8];
#pragma unroll
  for (int c2 = 0; c2 < 4; ++c2) {
    const int j = nb * 64 + c2 * 16 + nl;
    const float znv = zn[j], chv = ch[j], shv = sh[j];
#pragma unroll
    for (int r = 0; r < 8; ++r) {
      const int i = mb * 128 + w * 16 + r + 8 * h8;
      const float dot = acc[c2][r];
      const float arg = (2.f * RCC * dot * chv - (1.f + cr[r]) * shv)
                        / fmaxf(1.f - cr[r], 1e-15f);
      const float wv = 2.f * (znv / RCC) * asinhf(arg);
      yraw[(size_t)i * DD + j] = sinhf(RCC * wv) / RCC;
    }
  }
}

// ---------------------------------------------------------------------------
// K4: per-(b,h,s) head epilogue: project -> exp-ratio -> project
// mode 0: q [B,H,S,dh] + qx2 ; mode 1: kT [B,H,dh,S] + kx2 ; mode 2: gamma*v + gm1
// ---------------------------------------------------------------------------
__global__ __launch_bounds__(64)
void proj_epilogue(const float* __restrict__ yraw, int mode,
                   unsigned short* __restrict__ dst,
                   float* __restrict__ x2out, float* __restrict__ gm1out)
{
  const int idx = blockIdx.x;             // [0, B*S*H)
  const int row = idx >> 4;               // b*S + s
  const int h = idx & 15;
  const int b = row >> 10, s = row & 1023;
  const int t = threadIdx.x;              // 0..63
  const float y = yraw[(size_t)row * DD + h * DH + t];
  __shared__ float red[64];
  red[t] = y * y;
  __syncthreads();
  for (int o = 32; o; o >>= 1) {
    if (t < o) red[t] += red[t + o];
    __syncthreads();
  }
  const float s0 = red[0];
  const float norm = fmaxf(sqrtf(s0), 1e-15f);
  const float maxn = (1.f - PROJ_EPS) / RCC;
  const float f1 = (norm > maxn) ? maxn / norm : 1.f;
  const float s1 = (norm > maxn) ? maxn * maxn : s0;
  const float den = 1.f + sqrtf(1.f + RCC * RCC * s1);
  const float y2 = y * f1 / den;
  const float n2sq = s1 / (den * den);
  const float n2 = fmaxf(sqrtf(n2sq), 1e-15f);
  const float f2 = (n2 > maxn) ? maxn / n2 : 1.f;
  const float pt = y2 * f2;
  const float x2 = RCC * RCC * n2sq * f2 * f2;
  const int bh = b * NHEADS + h;
  if (mode == 1) {                                        // k, transposed [B,H,dh,S]
    dst[((size_t)bh * DH + t) * SS + s] = f2bf(pt);
    if (t == 0 && x2out) x2out[(size_t)bh * SS + s] = x2;
  } else if (mode == 2) {                                 // gamma*v
    const float g = 2.f / fmaxf(1.f - x2, 1e-15f);
    dst[((size_t)bh * SS + s) * DH + t] = f2bf(g * pt);
    if (t == 0 && gm1out) gm1out[(size_t)bh * SS + s] = g - 1.f;
  } else {                                                // q
    dst[((size_t)bh * SS + s) * DH + t] = f2bf(pt);
    if (t == 0 && x2out) x2out[(size_t)bh * SS + s] = x2;
  }
}

// ---------------------------------------------------------------------------
// K5: fused flash-style hyperbolic attention + gyromidpoint
// block: 128 thr (4 waves), one (b,h)+64-query block; key tiles of 32,
// double-buffered async K/V staging
// ---------------------------------------------------------------------------
__global__ __launch_bounds__(128)
void flash_attn(const unsigned short* __restrict__ qb,
                const unsigned short* __restrict__ kTb,
                const unsigned short* __restrict__ vgb,
                const float* __restrict__ qx2, const float* __restrict__ kx2,
                const float* __restrict__ gm1, const float* __restrict__ mask,
                float* __restrict__ out)
{
  __shared__ __align__(32) unsigned short Ks[2][64 * 32];    // [buf][kdim][key]
  __shared__ __align__(32) unsigned short Vs[2][32 * 64];    // [buf][key][d]
  __shared__ __align__(32) unsigned short Ps[4][16 * 32];    // per-wave probs tile
  __shared__ float k2s[2][32], g1s[2][32];
  const int qb_i = blockIdx.x & 15, bh = blockIdx.x >> 4;
  const int b = bh >> 4, h = bh & 15;
  const int tid = threadIdx.x, w = tid >> 5, lane = tid & 31;
  const int h8 = lane >> 4, nl = lane & 15;
  const int qstrip = qb_i * 64 + w * 16;
  const size_t bhS = (size_t)bh * SS;

  auto stageKV = [&](int jbase, int buf) {
#pragma unroll
    for (int c = 0; c < 2; ++c) {                            // K (64x32) + V (32x64)
      const int cc = tid + c * 128;
      const int krow = cc >> 2, kcg = cc & 3;
      async_copy16(&Ks[buf][krow * 32 + kcg * 8],
                   &kTb[((size_t)bh * DH + krow) * SS + jbase + kcg * 8]);
      const int vrow = cc >> 3, vcg = cc & 7;
      async_copy16(&Vs[buf][vrow * 64 + vcg * 8],
                   &vgb[(bhS + jbase + vrow) * DH + vcg * 8]);
    }
    if (tid < 32) k2s[buf][tid] = kx2[bhS + jbase + tid];
    else if (tid < 64) g1s[buf][tid - 32] = gm1[bhS + jbase + tid - 32];
  };

  Frag32 aq[2];
  {
    const unsigned short* qrow = qb + (bhS + qstrip + nl) * DH;
#pragma unroll
    for (int kk = 0; kk < 2; ++kk) {
      aq[kk].q[0] = *(const uint4*)&qrow[kk * 32 + h8 * 8];
      aq[kk].q[1] = *(const uint4*)&qrow[kk * 32 + h8 * 8 + 16];
    }
  }
  float q2r[8], rm[8], Lp[8], Dp[8];
#pragma unroll
  for (int r = 0; r < 8; ++r) {
    q2r[r] = qx2[bhS + qstrip + r + 8 * h8];
    rm[r] = -INFINITY; Lp[r] = 0.f; Dp[r] = 0.f;
  }
  v8f o[4] = {};

  stageKV(0, 0);
  async_join();
  __syncthreads();

  for (int jt = 0; jt < SS / 32; ++jt) {
    const int jbase = jt * 32;
    const int buf = jt & 1;
    if (jt + 1 < SS / 32) stageKV(jbase + 32, buf ^ 1);      // prefetch next tile (async)

    v8f sc[2] = {};
#pragma unroll
    for (int c = 0; c < 2; ++c)
#pragma unroll
      for (int kk = 0; kk < 2; ++kk) {
        v16bf bk;
#pragma unroll
        for (int s = 0; s < 16; ++s) {
          const int kx = (s < 8) ? h8 * 8 + s : 16 + h8 * 8 + (s - 8);
          bk[s] = __builtin_bit_cast(__bf16, Ks[buf][(kk * 32 + kx) * 32 + c * 16 + nl]);
        }
        sc[c] = __builtin_amdgcn_wmma_f32_16x16x32_bf16(
            false, aq[kk].b, false, bk, (short)0, sc[c], false, false);
      }
    float k2c[2], g1c[2], mkc[2];
#pragma unroll
    for (int c = 0; c < 2; ++c) {
      k2c[c] = k2s[buf][c * 16 + nl];
      g1c[c] = g1s[buf][c * 16 + nl];
      mkc[c] = mask[b * SS + jbase + c * 16 + nl];
    }
#pragma unroll
    for (int r = 0; r < 8; ++r) {
      float sr[2];
#pragma unroll
      for (int c = 0; c < 2; ++c) {
        const float dot = sc[c][r];
        const float num = fmaxf(q2r[r] - 2.f * dot + k2c[c], 1e-15f);
        const float dn = fmaxf((1.f - q2r[r]) * (1.f - k2c[c]), 1e-15f);
        sr[c] = -(1.f / RCC) * acoshf(1.f + 2.f * RCC * RCC * num / dn) + mkc[c];
      }
      const float tm = rmax16(fmaxf(sr[0], sr[1]));
      const float newm = fmaxf(rm[r], tm);
      const float fac = expf(rm[r] - newm);
      rm[r] = newm;
      const float p0 = expf(sr[0] - newm), p1 = expf(sr[1] - newm);
      Lp[r] = Lp[r] * fac + p0 + p1;
      Dp[r] = Dp[r] * fac + p0 * g1c[0] + p1 * g1c[1];
#pragma unroll
      for (int c2 = 0; c2 < 4; ++c2) o[c2][r] *= fac;
      Ps[w][(r + 8 * h8) * 32 + nl] = f2bf(p0);
      Ps[w][(r + 8 * h8) * 32 + 16 + nl] = f2bf(p1);
    }
    __syncthreads();                                         // Ps visible (wave-local tile)
    Frag32 ap;
    ap.q[0] = *(const uint4*)&Ps[w][nl * 32 + h8 * 8];
    ap.q[1] = *(const uint4*)&Ps[w][nl * 32 + h8 * 8 + 16];
#pragma unroll
    for (int c2 = 0; c2 < 4; ++c2) {
      v16bf bv;
#pragma unroll
      for (int s = 0; s < 16; ++s) {
        const int kx = (s < 8) ? h8 * 8 + s : 16 + h8 * 8 + (s - 8);
        bv[s] = __builtin_bit_cast(__bf16, Vs[buf][kx * 64 + c2 * 16 + nl]);
      }
      o[c2] = __builtin_amdgcn_wmma_f32_16x16x32_bf16(
          false, ap.b, false, bv, (short)0, o[c2], false, false);
    }
    async_join();                                            // next K/V tile landed
    __syncthreads();
  }

  const float maxn = (1.f - PROJ_EPS) / RCC;
#pragma unroll
  for (int r = 0; r < 8; ++r) {
    const float L = rsum16(Lp[r]);
    const float Dn = rsum16(Dp[r]);
    const float dq = fmaxf(Dn, 1e-10f * L);                  // == (Num/L)/max(Den/L,1e-10)
    float tmv[4], sqp = 0.f;
#pragma unroll
    for (int c2 = 0; c2 < 4; ++c2) { tmv[c2] = o[c2][r] / dq; sqp += tmv[c2] * tmv[c2]; }
    const float sq = rsum16(sqp);
    const float dfac = 1.f + sqrtf(fmaxf(1.f - RCC * RCC * sq, 1e-15f));
    const float n2sq = sq / (dfac * dfac);
    const float n2 = fmaxf(sqrtf(n2sq), 1e-15f);
    const float f2 = (n2 > maxn) ? maxn / n2 : 1.f;
    const int srow = qstrip + r + 8 * h8;
    float* orow = out + ((size_t)b * SS + srow) * DD + h * DH;
#pragma unroll
    for (int c2 = 0; c2 < 4; ++c2)
      orow[c2 * 16 + nl] = tmv[c2] / dfac * f2;
  }
}

// ---------------------------------------------------------------------------
extern "C" void kernel_launch(void* const* d_in, const int* in_sizes, int n_in,
                              void* d_out, int out_size, void* d_ws, size_t ws_size,
                              hipStream_t stream) {
  const float* hs  = (const float*)d_in[0];
  const float* msk = (const float*)d_in[1];
  const float* qz  = (const float*)d_in[2];
  const float* qr  = (const float*)d_in[3];
  const float* kz  = (const float*)d_in[4];
  const float* kr  = (const float*)d_in[5];
  const float* vz  = (const float*)d_in[6];
  const float* vr  = (const float*)d_in[7];
  float* out = (float*)d_out;

  char* p = (char*)d_ws;
  auto alloc = [&](size_t bytes) -> char* {
    char* r = p;
    p += (bytes + 255) & ~(size_t)255;
    return r;
  };
  unsigned short* xb   = (unsigned short*)alloc((size_t)BB * SS * DD * 2);
  unsigned short* zub  = (unsigned short*)alloc((size_t)3 * DD * DD * 2);
  float* zn   = (float*)alloc((size_t)3 * DD * 4);
  float* chA  = (float*)alloc((size_t)3 * DD * 4);
  float* shA  = (float*)alloc((size_t)3 * DD * 4);
  float* cx2  = (float*)alloc((size_t)BB * SS * 4);
  float* yraw = (float*)alloc((size_t)BB * SS * DD * 4);
  unsigned short* qbuf = (unsigned short*)alloc((size_t)BB * NHEADS * SS * DH * 2);
  unsigned short* kT   = (unsigned short*)alloc((size_t)BB * NHEADS * DH * SS * 2);
  unsigned short* vg   = (unsigned short*)alloc((size_t)BB * NHEADS * SS * DH * 2);
  float* qx2 = (float*)alloc((size_t)BB * NHEADS * SS * 4);
  float* kx2 = (float*)alloc((size_t)BB * NHEADS * SS * 4);
  float* gm1 = (float*)alloc((size_t)BB * NHEADS * SS * 4);

  prep_weights<<<3 * DD, 256, 0, stream>>>(qz, qr, kz, kr, vz, vr, zub, zn, chA, shA);
  prep_x<<<BB * SS, 256, 0, stream>>>(hs, xb, cx2);
  for (int pj = 0; pj < 3; ++pj) {
    proj_gemm<<<dim3(DD / 64, BB * SS / 128), 256, 0, stream>>>(
        xb, zub + (size_t)pj * DD * DD, zn + pj * DD, chA + pj * DD, shA + pj * DD,
        cx2, yraw);
    unsigned short* dst = (pj == 0) ? qbuf : ((pj == 1) ? kT : vg);
    float* x2o = (pj == 0) ? qx2 : ((pj == 1) ? kx2 : nullptr);
    proj_epilogue<<<BB * SS * NHEADS, 64, 0, stream>>>(
        yraw, pj, dst, x2o, (pj == 2) ? gm1 : nullptr);
  }
  flash_attn<<<BB * NHEADS * (SS / 64), 128, 0, stream>>>(
      qbuf, kT, vg, qx2, kx2, gm1, msk, out);
}
